// Qwen35GatedDeltaNet_2542620639627
// MI455X (gfx1250) — matrix operand order repaired and checked by reading, not compiled, not run on hardware
//
#include <hip/hip_runtime.h>

// ---------------- Problem constants ----------------
#define BATCH   2
#define SEQ     2048
#define DMODEL  2048
#define HV      32
#define HK      16
#define DK      128
#define DV      128
#define KCONV   4
#define KEY_DIM   (HK * DK)            // 2048
#define VAL_DIM   (HV * DV)            // 4096
#define CONV_DIM  (2 * KEY_DIM + VAL_DIM)  // 8192
#define ROWS      (BATCH * SEQ)        // 4096
#define NPAD      64                   // padded width for W_b / W_a GEMMs

typedef __bf16 v16bf __attribute__((ext_vector_type(16)));
typedef float  v8f   __attribute__((ext_vector_type(8)));

// =====================================================================
// f32 -> bf16 conversion (grid-stride)
// =====================================================================
__global__ void f32_to_bf16_kernel(const float* __restrict__ in,
                                   __bf16* __restrict__ out, size_t n) {
  size_t i = (size_t)blockIdx.x * blockDim.x + threadIdx.x;
  size_t stride = (size_t)gridDim.x * blockDim.x;
  for (; i < n; i += stride) out[i] = (__bf16)in[i];
}

// f32 [rows, cin] -> bf16 [rows, cout] zero-padded on columns
__global__ void f32_to_bf16_pad_kernel(const float* __restrict__ in,
                                       __bf16* __restrict__ out,
                                       int rows, int cin, int cout) {
  size_t total = (size_t)rows * cout;
  size_t i = (size_t)blockIdx.x * blockDim.x + threadIdx.x;
  size_t stride = (size_t)gridDim.x * blockDim.x;
  for (; i < total; i += stride) {
    int c = (int)(i % cout);
    int r = (int)(i / cout);
    out[i] = (c < cin) ? (__bf16)in[(size_t)r * cin + c] : (__bf16)0.0f;
  }
}

// =====================================================================
// Tiled bf16 WMMA GEMM: C[M,N](f32) = A[M,K](bf16,rowmaj) * B[K,N](bf16,rowmaj)
// REQUIRES: M%64==0, N%64==0, K%32==0 (call sites guarantee via padding).
// Block tile 64x64, 128 threads (4 waves), wave owns a 32x32 quadrant =
// 2x2 v_wmma_f32_16x16x32_bf16 tiles. Register double-buffered staging:
// next K-chunk's global loads are issued while the current chunk's WMMAs
// run, so loadcnt waits overlap compute. A staged row-major, B staged
// TRANSPOSED in LDS so every fragment is two aligned ds_load_b128.
// =====================================================================
#define TM 64
#define TN 64
#define TK 32

__global__ __launch_bounds__(128)
void gemm_bf16_wmma(const __bf16* __restrict__ A, const __bf16* __restrict__ Bm,
                    float* __restrict__ C, int M, int N, int K) {
  __shared__ __align__(16) __bf16 As[TM][TK + 8];   // 80B row stride (16B mult)
  __shared__ __align__(16) __bf16 Bt[TN][TK + 8];   // transposed: Bt[n][k]

  const int tid  = threadIdx.x;
  const int lane = tid & 31;
  const int wave = tid >> 5;
  const int hlf  = (lane >> 4) & 1;   // lane half (0: lanes 0-15, 1: 16-31)
  const int l16  = lane & 15;

  const int blockM = blockIdx.y * TM;
  const int blockN = blockIdx.x * TN;
  const int wm = (wave & 1) * 32;     // wave quadrant inside block tile
  const int wn = (wave >> 1) * 32;

  // per-thread staging coordinates (2 A-chunks + 2 B-chunks of 16B each)
  const int ar0 = tid >> 2;           // A rows ar0 and ar0+32
  const int ac8 = (tid & 3) * 8;
  const int br0 = tid >> 3;           // B k-rows br0 and br0+16
  const int bn8 = (tid & 7) * 8;

  const __bf16* aSrc0 = A + (size_t)(blockM + ar0) * K + ac8;
  const __bf16* aSrc1 = A + (size_t)(blockM + ar0 + 32) * K + ac8;
  const __bf16* bSrc0 = Bm + (size_t)br0 * N + blockN + bn8;
  const __bf16* bSrc1 = Bm + (size_t)(br0 + 16) * N + blockN + bn8;

  v8f acc[2][2];
#pragma unroll
  for (int im = 0; im < 2; ++im)
#pragma unroll
    for (int in = 0; in < 2; ++in)
#pragma unroll
      for (int r = 0; r < 8; ++r) acc[im][in][r] = 0.0f;

  // prologue: fetch first K-chunk into registers
  uint4 aReg0 = *reinterpret_cast<const uint4*>(aSrc0);
  uint4 aReg1 = *reinterpret_cast<const uint4*>(aSrc1);
  uint4 bReg0 = *reinterpret_cast<const uint4*>(bSrc0);
  uint4 bReg1 = *reinterpret_cast<const uint4*>(bSrc1);

  for (int k0 = 0; k0 < K; k0 += TK) {
    // ---- drain staged registers into LDS ----
    *reinterpret_cast<uint4*>(&As[ar0][ac8])      = aReg0;
    *reinterpret_cast<uint4*>(&As[ar0 + 32][ac8]) = aReg1;
    {
      const __bf16* e = reinterpret_cast<const __bf16*>(&bReg0);
#pragma unroll
      for (int j = 0; j < 8; ++j) Bt[bn8 + j][br0] = e[j];
    }
    {
      const __bf16* e = reinterpret_cast<const __bf16*>(&bReg1);
#pragma unroll
      for (int j = 0; j < 8; ++j) Bt[bn8 + j][br0 + 16] = e[j];
    }
    __syncthreads();

    // ---- issue next chunk's global loads (overlap with WMMA below) ----
    if (k0 + TK < K) {
      aSrc0 += TK;
      aSrc1 += TK;
      bSrc0 += (size_t)TK * N;
      bSrc1 += (size_t)TK * N;
      aReg0 = *reinterpret_cast<const uint4*>(aSrc0);
      aReg1 = *reinterpret_cast<const uint4*>(aSrc1);
      bReg0 = *reinterpret_cast<const uint4*>(bSrc0);
      bReg1 = *reinterpret_cast<const uint4*>(bSrc1);
    }

    // ---- fragments (contiguous k order, same in A and B) ----
    // A 16x32: frag[e] = A[m][K = 16*hlf + e] (contiguous, matches B)
    v16bf afrag[2], bfrag[2];
#pragma unroll
    for (int im = 0; im < 2; ++im) {
      const int mRow = wm + im * 16 + l16;
#pragma unroll
      for (int e = 0; e < 8; ++e) afrag[im][e] = As[mRow][hlf * 16 + e];
#pragma unroll
      for (int e = 0; e < 8; ++e) afrag[im][8 + e] = As[mRow][hlf * 16 + 8 + e];
    }
    // B 32x16: frag[e] = B[K = 16*hlf + e][n] -> contiguous in Bt[n][...]
#pragma unroll
    for (int in = 0; in < 2; ++in) {
      const int nCol = wn + in * 16 + l16;
#pragma unroll
      for (int e = 0; e < 16; ++e) bfrag[in][e] = Bt[nCol][hlf * 16 + e];
    }

#pragma unroll
    for (int im = 0; im < 2; ++im)
#pragma unroll
      for (int in = 0; in < 2; ++in)
        acc[im][in] = __builtin_amdgcn_wmma_f32_16x16x32_bf16(
            false, afrag[im], false, bfrag[in], (short)0, acc[im][in],
            false, false);
    __syncthreads();
  }

  // ---- store D: vgpr r holds M = r + 8*hlf, lane l16 holds N ----
#pragma unroll
  for (int im = 0; im < 2; ++im)
#pragma unroll
    for (int in = 0; in < 2; ++in)
#pragma unroll
      for (int r = 0; r < 8; ++r) {
        const int m = blockM + wm + im * 16 + hlf * 8 + r;
        const int n = blockN + wn + in * 16 + l16;
        C[(size_t)m * N + n] = acc[im][in][r];
      }
}

// =====================================================================
// Causal depthwise conv1d (K=4, left-pad 3) + SiLU, routed into q/k/v
// =====================================================================
__global__ void conv_silu_route_kernel(const float* __restrict__ mixed_pre,
                                       const float* __restrict__ conv_w,
                                       float* __restrict__ qraw,
                                       float* __restrict__ kraw,
                                       float* __restrict__ vbuf) {
  const size_t total = (size_t)ROWS * CONV_DIM;
  size_t idx = (size_t)blockIdx.x * blockDim.x + threadIdx.x;
  const size_t stride = (size_t)gridDim.x * blockDim.x;
  for (; idx < total; idx += stride) {
    const int c = (int)(idx % CONV_DIM);
    const size_t bl = idx / CONV_DIM;        // b*SEQ + l
    const int l = (int)(bl % SEQ);
    float acc = 0.0f;
#pragma unroll
    for (int j = 0; j < KCONV; ++j) {
      const int li = l + j - (KCONV - 1);
      if (li >= 0)
        acc += conv_w[c * KCONV + j] *
               mixed_pre[(bl + (size_t)(j - (KCONV - 1))) * CONV_DIM + c];
    }
    const float y = acc / (1.0f + __expf(-acc));   // SiLU
    if (c < KEY_DIM)                qraw[bl * KEY_DIM + c] = y;
    else if (c < 2 * KEY_DIM)       kraw[bl * KEY_DIM + (c - KEY_DIM)] = y;
    else                            vbuf[bl * VAL_DIM + (c - 2 * KEY_DIM)] = y;
  }
}

// =====================================================================
// Per-head L2 norm over 128 dims (in-place), then * scale
// =====================================================================
__global__ __launch_bounds__(128)
void l2norm_scale_kernel(float* __restrict__ x, float scale) {
  __shared__ float red[128];
  const int t = threadIdx.x;
  const size_t base = (size_t)blockIdx.x * 128;
  const float v = x[base + t];
  red[t] = v * v;
  __syncthreads();
#pragma unroll
  for (int s = 64; s > 0; s >>= 1) {
    if (t < s) red[t] += red[t + s];
    __syncthreads();
  }
  const float rs = rsqrtf(red[0] + 1e-6f) * scale;
  x[base + t] = v * rs;
}

// =====================================================================
// beta = sigmoid(b); g = -exp(A_log) * softplus(a + dt_bias)
// bvec/avec live in padded [ROWS, NPAD] GEMM outputs.
// =====================================================================
__global__ void gate_kernel(const float* __restrict__ bvec,
                            const float* __restrict__ avec,
                            const float* __restrict__ dt_bias,
                            const float* __restrict__ A_log,
                            float* __restrict__ beta,
                            float* __restrict__ gdec) {
  const size_t total = (size_t)ROWS * HV;
  size_t i = (size_t)blockIdx.x * blockDim.x + threadIdx.x;
  const size_t stride = (size_t)gridDim.x * blockDim.x;
  for (; i < total; i += stride) {
    const int h = (int)(i % HV);
    const size_t row = i / HV;
    const size_t src = row * NPAD + h;
    beta[i] = 1.0f / (1.0f + __expf(-bvec[src]));
    const float xv = avec[src] + dt_bias[h];
    const float sp = (xv > 20.0f) ? xv : log1pf(__expf(xv));
    gdec[i] = -__expf(A_log[h]) * sp;
  }
}

// =====================================================================
// Gated delta-rule scan. One block per (b,h) state; thread t owns S[:,t].
// S kept fully in VGPRs (128 floats per lane).
// =====================================================================
__global__ __launch_bounds__(128)
void delta_scan_kernel(const float* __restrict__ qn,
                       const float* __restrict__ kn,
                       const float* __restrict__ vbuf,
                       const float* __restrict__ gdec,
                       const float* __restrict__ beta,
                       float* __restrict__ core) {
  const int b  = blockIdx.x / HV;
  const int h  = blockIdx.x % HV;
  const int hk = h >> 1;                 // HV/HK == 2 (GQA repeat)
  const int t  = threadIdx.x;            // DV column

  __shared__ float kS[DK];
  __shared__ float qS[DK];
  __shared__ float sc[2];

  float S[DK];
#pragma unroll
  for (int r = 0; r < DK; ++r) S[r] = 0.0f;

  for (int l = 0; l < SEQ; ++l) {
    const size_t bl = (size_t)b * SEQ + l;
    kS[t] = kn[(bl * HK + hk) * DK + t];
    qS[t] = qn[(bl * HK + hk) * DK + t];
    const float v = vbuf[(bl * HV + h) * DV + t];
    if (t == 0) {
      sc[0] = __expf(gdec[bl * HV + h]);
      sc[1] = beta[bl * HV + h];
    }
    __syncthreads();
    const float eg = sc[0], bt = sc[1];

    float kv = 0.0f;
#pragma unroll
    for (int r = 0; r < DK; ++r) kv += kS[r] * S[r];
    const float delta = (v - kv * eg) * bt;

    float o = 0.0f;
#pragma unroll
    for (int r = 0; r < DK; ++r) {
      S[r] = S[r] * eg + kS[r] * delta;
      o += qS[r] * S[r];
    }
    core[(bl * HV + h) * DV + t] = o;
    __syncthreads();
  }
}

// =====================================================================
// gated = core * silu(z); RMS-normalize over DV; * norm_w; emit bf16
// =====================================================================
__global__ __launch_bounds__(128)
void gated_rmsnorm_kernel(const float* __restrict__ core,
                          const float* __restrict__ zbuf,
                          const float* __restrict__ norm_w,
                          __bf16* __restrict__ gated) {
  __shared__ float red[128];
  const int t = threadIdx.x;
  const size_t base = (size_t)blockIdx.x * DV + t;
  const float z = zbuf[base];
  const float x = core[base] * (z / (1.0f + __expf(-z)));
  red[t] = x * x;
  __syncthreads();
#pragma unroll
  for (int s = 64; s > 0; s >>= 1) {
    if (t < s) red[t] += red[t + s];
    __syncthreads();
  }
  const float ms = red[0] * (1.0f / (float)DV);
  gated[base] = (__bf16)(x * rsqrtf(ms + 1e-6f) * norm_w[t]);
}

// =====================================================================
// Host side
// =====================================================================
static inline void launch_gemm(const __bf16* A, const __bf16* Bm, float* C,
                               int M, int N, int K, hipStream_t s) {
  dim3 grid((N + TN - 1) / TN, (M + TM - 1) / TM);
  gemm_bf16_wmma<<<grid, 128, 0, s>>>(A, Bm, C, M, N, K);
}

static inline void launch_cvt(const float* in, __bf16* out, size_t n,
                              hipStream_t s) {
  int blocks = (int)((n + 255) / 256);
  if (blocks > 4096) blocks = 4096;
  f32_to_bf16_kernel<<<blocks, 256, 0, s>>>(in, out, n);
}

extern "C" void kernel_launch(void* const* d_in, const int* in_sizes, int n_in,
                              void* d_out, int out_size, void* d_ws, size_t ws_size,
                              hipStream_t stream) {
  (void)in_sizes; (void)n_in; (void)out_size; (void)ws_size;

  const float* hidden  = (const float*)d_in[0];   // [B,L,D]
  const float* W_qkv   = (const float*)d_in[1];   // [D, CONV_DIM]
  const float* W_z     = (const float*)d_in[2];   // [D, VAL_DIM]
  const float* W_b     = (const float*)d_in[3];   // [D, HV]
  const float* W_a     = (const float*)d_in[4];   // [D, HV]
  const float* conv_w  = (const float*)d_in[5];   // [CONV_DIM, KCONV]
  const float* dt_bias = (const float*)d_in[6];   // [HV]
  const float* A_log   = (const float*)d_in[7];   // [HV]
  const float* norm_w  = (const float*)d_in[8];   // [DV]
  const float* W_out   = (const float*)d_in[9];   // [VAL_DIM, D]
  float* out = (float*)d_out;                     // [B,L,D]

  // ---- workspace carve-out ----
  char* ws = (char*)d_ws;
  size_t off = 0;
  auto carve = [&](size_t bytes) -> void* {
    void* p = ws + off;
    off += (bytes + 255) & ~(size_t)255;
    return p;
  };
  __bf16* hb     = (__bf16*)carve((size_t)ROWS * DMODEL * 2);
  __bf16* wqkvb  = (__bf16*)carve((size_t)DMODEL * CONV_DIM * 2);
  __bf16* wzb    = (__bf16*)carve((size_t)DMODEL * VAL_DIM * 2);
  __bf16* wbb    = (__bf16*)carve((size_t)DMODEL * NPAD * 2);   // padded 32->64
  __bf16* wab    = (__bf16*)carve((size_t)DMODEL * NPAD * 2);   // padded 32->64
  __bf16* woutb  = (__bf16*)carve((size_t)VAL_DIM * DMODEL * 2);
  float* mixed   = (float*)carve((size_t)ROWS * CONV_DIM * 4);
  float* qraw    = (float*)carve((size_t)ROWS * KEY_DIM * 4);
  float* kraw    = (float*)carve((size_t)ROWS * KEY_DIM * 4);
  float* vbuf    = (float*)carve((size_t)ROWS * VAL_DIM * 4);
  float* zbuf    = (float*)carve((size_t)ROWS * VAL_DIM * 4);
  float* bvec    = (float*)carve((size_t)ROWS * NPAD * 4);      // padded
  float* avec    = (float*)carve((size_t)ROWS * NPAD * 4);      // padded
  float* betab   = (float*)carve((size_t)ROWS * HV * 4);
  float* gdec    = (float*)carve((size_t)ROWS * HV * 4);
  float* corebuf = (float*)carve((size_t)ROWS * VAL_DIM * 4);
  __bf16* gated  = (__bf16*)carve((size_t)ROWS * VAL_DIM * 2);

  // 1) convert operands to bf16 for the WMMA path (W_b/W_a zero-padded to 64)
  launch_cvt(hidden, hb,    (size_t)ROWS * DMODEL, stream);
  launch_cvt(W_qkv,  wqkvb, (size_t)DMODEL * CONV_DIM, stream);
  launch_cvt(W_z,    wzb,   (size_t)DMODEL * VAL_DIM, stream);
  launch_cvt(W_out,  woutb, (size_t)VAL_DIM * DMODEL, stream);
  {
    size_t total = (size_t)DMODEL * NPAD;
    int blocks = (int)((total + 255) / 256);
    f32_to_bf16_pad_kernel<<<blocks, 256, 0, stream>>>(W_b, wbb, DMODEL, HV, NPAD);
    f32_to_bf16_pad_kernel<<<blocks, 256, 0, stream>>>(W_a, wab, DMODEL, HV, NPAD);
  }

  // 2) projections (WMMA bf16, f32 accumulate) — all full-tile
  launch_gemm(hb, wqkvb, mixed, ROWS, CONV_DIM, DMODEL, stream);
  launch_gemm(hb, wzb,   zbuf,  ROWS, VAL_DIM,  DMODEL, stream);
  launch_gemm(hb, wbb,   bvec,  ROWS, NPAD,     DMODEL, stream);
  launch_gemm(hb, wab,   avec,  ROWS, NPAD,     DMODEL, stream);

  // 3) causal depthwise conv + SiLU, route q/k/v
  {
    size_t total = (size_t)ROWS * CONV_DIM;
    int blocks = (int)((total + 255) / 256);
    if (blocks > 65536) blocks = 65536;
    conv_silu_route_kernel<<<blocks, 256, 0, stream>>>(mixed, conv_w, qraw,
                                                       kraw, vbuf);
  }

  // 4) per-head L2 norm (q also * DK^-0.5); gates
  l2norm_scale_kernel<<<ROWS * HK, 128, 0, stream>>>(qraw, 0.08838834764831845f);
  l2norm_scale_kernel<<<ROWS * HK, 128, 0, stream>>>(kraw, 1.0f);
  {
    size_t total = (size_t)ROWS * HV;
    int blocks = (int)((total + 255) / 256);
    gate_kernel<<<blocks, 256, 0, stream>>>(bvec, avec, dt_bias, A_log, betab,
                                            gdec);
  }

  // 5) gated delta-rule scan (one block per (b,h) state)
  delta_scan_kernel<<<BATCH * HV, 128, 0, stream>>>(qraw, kraw, vbuf, gdec,
                                                    betab, corebuf);

  // 6) gated RMSNorm -> bf16
  gated_rmsnorm_kernel<<<ROWS * HV, 128, 0, stream>>>(corebuf, zbuf, norm_w,
                                                      gated);

  // 7) output projection (WMMA bf16 -> f32 out)
  launch_gemm(gated, woutb, out, ROWS, DMODEL, VAL_DIM, stream);
}
